// GPTNet_63136019251512
// MI455X (gfx1250) — matrix operand order repaired and checked by reading, not compiled, not run on hardware
//
#include <hip/hip_runtime.h>
#include <hip/hip_bf16.h>
#include <math.h>

typedef __attribute__((ext_vector_type(16))) __bf16 bf16x16;
typedef __attribute__((ext_vector_type(8)))  __bf16 bf16x8;
typedef __attribute__((ext_vector_type(8)))  float  floatx8;

#define NB   128
#define TT   128
#define VV   25
#define TV   3200          // TT*VV
#define CI_  16

__device__ __forceinline__ float lrelu_f(float x) { return x >= 0.f ? x : 0.1f * x; }

// branch-free tanh: clamp + v_exp_f32 + v_rcp_f32 (no IEEE div sequence)
__device__ __forceinline__ float tanh_fast(float x) {
    x = fminf(fmaxf(x, -10.f), 10.f);
    const float e = __expf(2.f * x);
    return (e - 1.f) * __builtin_amdgcn_rcpf(e + 1.f);
}

// ---------------------------------------------------------------------------
// Fused GEMM:  out[n,o,col] = act( res + bn( W(o,:) . in(n,:,col) + bias ) )
// KTAPS==1: 1x1 conv (K = CIN).  KTAPS==7: temporal conv (im2col'd at stage).
// Block = COUT/16 waves; stages ONE 16-column input tile into LDS (bf16,
// column-major) so each input element is read from HBM exactly once, then each
// wave computes a 16(M)x16(col) tile with bf16 WMMA, f32 accumulation.
// ---------------------------------------------------------------------------
template<int CIN, int COUT, int KTAPS, bool BN, bool RES>
__global__ void k_gemm(const float* __restrict__ in, const float* __restrict__ W,
                       const float* __restrict__ bias, const float* __restrict__ bn,
                       const float* __restrict__ res, float* __restrict__ out) {
    constexpr int Ktot = CIN * KTAPS;
    constexpr int NW   = COUT / 16;
    __shared__ __align__(16) __bf16 ldsT[16 * Ktot];   // [col][K], <=14KB

    const int n    = blockIdx.z;
    const int col0 = blockIdx.x * 16;
    const int tid  = threadIdx.x;               // 32*NW threads
    const float* inN = in + (size_t)n * CIN * TV;

    // ---- stage input tile (im2col for KTAPS>1), fp32 -> bf16, once per block
    for (int idx = tid; idx < 16 * Ktot; idx += 32 * NW) {
        const int col = idx & 15, Kb = idx >> 4;
        float val;
        if (KTAPS == 1) {
            val = inN[(size_t)Kb * TV + col0 + col];
        } else {
            const int i = Kb / KTAPS, k = Kb - i * KTAPS;
            const int gcol = col0 + col + (k - 3) * VV;   // == (t+k-3)*VV + v
            const int gc = gcol < 0 ? 0 : (gcol >= TV ? TV - 1 : gcol);
            val = inN[(size_t)i * TV + gc];
            val = (gcol == gc) ? val : 0.f;               // mask, no branch
        }
        ldsT[col * Ktot + Kb] = (__bf16)val;
    }
    __syncthreads();

    const int wave = tid >> 5, lane = tid & 31;
    const int half = lane >> 4, l15 = lane & 15;
    const int m0   = wave * 16;
    const int col  = col0 + l15;
    const float* wrow = W + (size_t)(m0 + l15) * Ktot;
    const __bf16* bcol = &ldsT[l15 * Ktot];

    floatx8 acc = {0.f,0.f,0.f,0.f,0.f,0.f,0.f,0.f};
    for (int k0 = 0; k0 < Ktot; k0 += 32) {
        // A: rows of W, two 8-float runs -> vectorized global loads
        const float4 wa0 = *(const float4*)&wrow[k0 + half * 8];
        const float4 wa1 = *(const float4*)&wrow[k0 + half * 8 + 4];
        const float4 wa2 = *(const float4*)&wrow[k0 + 16 + half * 8];
        const float4 wa3 = *(const float4*)&wrow[k0 + 16 + half * 8 + 4];
        bf16x16 a;
        a[0]=(__bf16)wa0.x; a[1]=(__bf16)wa0.y; a[2]=(__bf16)wa0.z; a[3]=(__bf16)wa0.w;
        a[4]=(__bf16)wa1.x; a[5]=(__bf16)wa1.y; a[6]=(__bf16)wa1.z; a[7]=(__bf16)wa1.w;
        a[8]=(__bf16)wa2.x; a[9]=(__bf16)wa2.y; a[10]=(__bf16)wa2.z; a[11]=(__bf16)wa2.w;
        a[12]=(__bf16)wa3.x; a[13]=(__bf16)wa3.y; a[14]=(__bf16)wa3.z; a[15]=(__bf16)wa3.w;
        // B: two 8-bf16 runs along K from LDS (same K order as A) -> 2x ds_load_b128
        const bf16x8 blo = *(const bf16x8*)&bcol[k0 + half * 8];
        const bf16x8 bhi = *(const bf16x8*)&bcol[k0 + 16 + half * 8];
        bf16x16 b;
#pragma unroll
        for (int e = 0; e < 8; ++e) { b[e] = blo[e]; b[8 + e] = bhi[e]; }
        acc = __builtin_amdgcn_wmma_f32_16x16x32_bf16(false, a, false, b, (short)0,
                                                      acc, false, false);
    }
#pragma unroll
    for (int r = 0; r < 8; ++r) {
        const int o = m0 + half * 8 + r;
        float val = acc[r] + bias[o];
        if (BN) {
            const float g = bn[o], bb = bn[COUT + o], mm = bn[2 * COUT + o],
                        vv2 = bn[3 * COUT + o];
            val = (val - mm) * (g * rsqrtf(vv2 + 1e-5f)) + bb;
        }
        if (RES) {
            val = lrelu_f(res[(size_t)n * COUT * TV + (size_t)o * TV + col] + val);
        }
        out[(size_t)n * COUT * TV + (size_t)o * TV + col] = val;
    }
}

// ---------------------------------------------------------------------------
// Spatial attention (25x25, K = CI*T = 2048): VALU with LDS chunking.
// ---------------------------------------------------------------------------
__global__ void k_attn_s(const float* __restrict__ qk, const float* __restrict__ alphas,
                         const float* __restrict__ att0s, float* __restrict__ attn) {
    const int ns = blockIdx.x;        // n*3 + s
    const int n = ns / 3, s = ns - n * 3;
    __shared__ float qs[64 * 25], ks2[64 * 25];
    const int tid = threadIdx.x;      // 256 threads
    float acc0 = 0.f, acc1 = 0.f, acc2 = 0.f;
    const float* qb = qk + (size_t)n * 96 * TV + (size_t)(s * CI_) * TV;
    const float* kb = qk + (size_t)n * 96 * TV + (size_t)((3 + s) * CI_) * TV;
    for (int j = 0; j < 2048; j += 64) {
        for (int idx = tid; idx < 64 * 25; idx += 256) {
            const int l = idx / 25, u = idx - l * 25;
            const int kk = j + l, c = kk >> 7, tt = kk & 127;
            qs[idx]  = qb[(size_t)c * TV + tt * VV + u];
            ks2[idx] = kb[(size_t)c * TV + tt * VV + u];
        }
        __syncthreads();
        for (int p = tid, pi = 0; p < 625; p += 256, ++pi) {
            const int u = p / 25, v = p - u * 25;
            float a = 0.f;
#pragma unroll 8
            for (int l = 0; l < 64; ++l) a += qs[l * 25 + u] * ks2[l * 25 + v];
            if (pi == 0) acc0 += a; else if (pi == 1) acc1 += a; else acc2 += a;
        }
        __syncthreads();
    }
    for (int p = tid, pi = 0; p < 625; p += 256, ++pi) {
        const float a = (pi == 0 ? acc0 : pi == 1 ? acc1 : acc2);
        attn[(size_t)ns * 625 + p] = tanh_fast(a * (1.f / 2048.f)) * alphas[s]
                                     + att0s[s * 625 + p];
    }
}

// y[n, s*64+c, t, v] = sum_u x[n,c,t,u] * attn[n,s,u,v]   (K=25, VALU)
__global__ void k_apply_s(const float* __restrict__ x, const float* __restrict__ attn,
                          float* __restrict__ y) {
    const int n = blockIdx.y;
    __shared__ float at[3 * 625];
    for (int i = threadIdx.x; i < 1875; i += 256) at[i] = attn[(size_t)n * 1875 + i];
    __syncthreads();
    const int row = blockIdx.x * 256 + threadIdx.x;   // (s,c,t), 24576 rows
    const int s = row / 8192, rem = row - s * 8192, c = rem >> 7, t = rem & 127;
    float xr[25];
    const float* xp = x + (size_t)n * 64 * TV + (size_t)c * TV + t * VV;
#pragma unroll
    for (int u = 0; u < 25; ++u) xr[u] = xp[u];
    float* yp = y + (size_t)n * 192 * TV + (size_t)(s * 64 + c) * TV + t * VV;
    const float* ap = at + s * 625;
#pragma unroll
    for (int v = 0; v < 25; ++v) {
        float acc = 0.f;
#pragma unroll
        for (int u = 0; u < 25; ++u) acc += xr[u] * ap[u * 25 + v];
        yp[v] = acc;
    }
}

// ---------------------------------------------------------------------------
// Temporal attention (T x T, K = CI*V = 400 padded to 416) via WMMA.
// Tiles staged to LDS as bf16 (coalesced, K-fastest), fragments via ds b128.
// qk layout (n, 8, 16, T, V): groups 0..1 q_f, 2..3 q_b, 4..5 k_f, 6..7 k_b
// ns in grid.z so co-resident blocks share one batch slice (L2-resident).
// ---------------------------------------------------------------------------
__global__ void k_attn_t(const float* __restrict__ qk, const float* __restrict__ alpf,
                         const float* __restrict__ alpb, float* __restrict__ a_f,
                         float* __restrict__ a_b) {
    const int t0 = blockIdx.x * 16, q0 = blockIdx.y * 16;
    const int ns = blockIdx.z;        // n*2 + s
    const int n = ns >> 1, s = ns & 1;
    const int lane = threadIdx.x, half = lane >> 4, l15 = lane & 15;
    const size_t base = (size_t)n * 128 * TV;
    __shared__ __align__(16) __bf16 qT[16 * 416];
    __shared__ __align__(16) __bf16 kT[16 * 416];

    for (int dir = 0; dir < 2; ++dir) {
        const int gq = (dir ? 2 : 0) + s, gk = (dir ? 6 : 4) + s;
        // stage 16x400 Q/K tiles (K-fastest -> coalesced 25-elem runs) + pad
        for (int idx = lane; idx < 16 * 400; idx += 32) {
            const int row = idx / 400, K = idx - row * 400;
            const int c = K / 25, v = K - c * 25;
            qT[row * 416 + K] = (__bf16)qk[base + (size_t)(gq * 16 + c) * TV
                                           + (t0 + row) * VV + v];
            kT[row * 416 + K] = (__bf16)qk[base + (size_t)(gk * 16 + c) * TV
                                           + (q0 + row) * VV + v];
        }
        for (int idx = lane; idx < 16 * 16; idx += 32) {
            const int row = idx >> 4, K = 400 + (idx & 15);
            qT[row * 416 + K] = (__bf16)0.f;
            kT[row * 416 + K] = (__bf16)0.f;
        }
        __syncthreads();

        const __bf16* qrow = &qT[l15 * 416];
        const __bf16* krow = &kT[l15 * 416];
        floatx8 acc = {0.f,0.f,0.f,0.f,0.f,0.f,0.f,0.f};
        for (int k0 = 0; k0 < 416; k0 += 32) {
            const bf16x8 a0 = *(const bf16x8*)&qrow[k0 + half * 8];
            const bf16x8 a1 = *(const bf16x8*)&qrow[k0 + 16 + half * 8];
            const bf16x8 b0 = *(const bf16x8*)&krow[k0 + half * 8];
            const bf16x8 b1 = *(const bf16x8*)&krow[k0 + 16 + half * 8];
            bf16x16 a, b;
#pragma unroll
            for (int e = 0; e < 8; ++e) {
                a[e] = a0[e]; a[8 + e] = a1[e];
                b[e] = b0[e]; b[8 + e] = b1[e];
            }
            acc = __builtin_amdgcn_wmma_f32_16x16x32_bf16(false, a, false, b, (short)0,
                                                          acc, false, false);
        }
        const float alpha = dir ? alpb[s] : alpf[s];
        float* dst = dir ? a_b : a_f;
#pragma unroll
        for (int r = 0; r < 8; ++r) {
            const int t = t0 + half * 8 + r, q = q0 + l15;
            const float m = dir ? (q >= t ? 1.f : 0.f) : (q <= t ? 1.f : 0.f);
            dst[((size_t)ns * TT + t) * TT + q] = tanh_fast(acc[r] * (1.f / 400.f)) * alpha * m;
        }
        __syncthreads();
    }
}

// z[n, dir*128 + s*64 + c, q, v] = sum_t y[n,c,t,v] * a[n,s,t,q]   (K = T = 128, WMMA)
__global__ void k_apply_t(const float* __restrict__ y, const float* __restrict__ a_f,
                          const float* __restrict__ a_b, float* __restrict__ z) {
    const int m0 = blockIdx.x * 16;     // rows = flattened (c,v), 1600 total
    const int q0 = blockIdx.y * 16;
    const int ns = blockIdx.z; const int n = ns >> 1, s = ns & 1;
    const int lane = threadIdx.x, half = lane >> 4, l15 = lane & 15;
    // A fragments (depend only on y) loaded once, reused for both directions
    bf16x16 afr[4];
    {
        const int rowA = m0 + l15, ca = rowA / 25, va = rowA - ca * 25;
        const float* yrow = y + (size_t)n * 64 * TV + (size_t)ca * TV + va;
#pragma unroll
        for (int ks = 0; ks < 4; ++ks)
#pragma unroll
            for (int e = 0; e < 16; ++e) {
                const int K = ks * 32 + (e < 8 ? half * 8 + e : 16 + half * 8 + (e - 8));
                afr[ks][e] = (__bf16)yrow[(size_t)K * VV];
            }
    }
    for (int dir = 0; dir < 2; ++dir) {
        const float* am = (dir ? a_b : a_f) + (size_t)ns * TT * TT;
        floatx8 acc = {0.f,0.f,0.f,0.f,0.f,0.f,0.f,0.f};
#pragma unroll
        for (int ks = 0; ks < 4; ++ks) {
            bf16x16 b;
#pragma unroll
            for (int e = 0; e < 16; ++e) {
                const int K = ks * 32 + (e < 8 ? half * 8 + e : 16 + half * 8 + (e - 8));
                b[e] = (__bf16)am[(size_t)K * TT + q0 + l15];
            }
            acc = __builtin_amdgcn_wmma_f32_16x16x32_bf16(false, afr[ks], false, b,
                                                          (short)0, acc, false, false);
        }
#pragma unroll
        for (int r = 0; r < 8; ++r) {
            const int row = m0 + half * 8 + r, c = row / 25, v = row - c * 25;
            const int q = q0 + l15;
            z[(size_t)n * 256 * TV + (size_t)(dir * 128 + s * 64 + c) * TV
              + (size_t)q * VV + v] = acc[r];
        }
    }
}

// ---------------------------------------------------------------------------
extern "C" void kernel_launch(void* const* d_in, const int* in_sizes, int n_in,
                              void* d_out, int out_size, void* d_ws, size_t ws_size,
                              hipStream_t stream) {
    (void)in_sizes; (void)n_in; (void)out_size; (void)ws_size;
    const float* x        = (const float*)d_in[0];
    const float* w_in_s   = (const float*)d_in[1];
    const float* b_in_s   = (const float*)d_in[2];
    const float* alphas   = (const float*)d_in[3];
    const float* att0s    = (const float*)d_in[4];
    const float* w_out_s  = (const float*)d_in[5];
    const float* b_out_s  = (const float*)d_in[6];
    const float* bn_out_s = (const float*)d_in[7];
    const float* w_ff_s   = (const float*)d_in[8];
    const float* b_ff_s   = (const float*)d_in[9];
    const float* bn_ff_s  = (const float*)d_in[10];
    const float* w_in_t   = (const float*)d_in[11];
    const float* b_in_t   = (const float*)d_in[12];
    const float* alphat_f = (const float*)d_in[13];
    const float* alphat_b = (const float*)d_in[14];
    const float* w_out_t  = (const float*)d_in[15];
    const float* b_out_t  = (const float*)d_in[16];
    const float* bn_out_t = (const float*)d_in[17];
    const float* w_ff_t   = (const float*)d_in[18];
    const float* b_ff_t   = (const float*)d_in[19];
    const float* bn_ff_t  = (const float*)d_in[20];
    const float* w_tcn    = (const float*)d_in[21];
    const float* b_tcn    = (const float*)d_in[22];
    const float* bn_tcn   = (const float*)d_in[23];
    float* out = (float*)d_out;

    // workspace layout (floats)
    float* ws   = (float*)d_ws;
    float* qkb  = ws;                              // 128ch buffer (qk_s/qk_t, later z)
    float* big  = ws + (size_t)52428800;           // 256ch buffer (spatial y / z)
    float* yf   = ws + (size_t)157286400;          // 64ch: y after spatial stage
    float* attn = ws + (size_t)183500800;          // spatial attention
    float* af   = ws + (size_t)183740800;          // temporal a_f
    float* ab   = ws + (size_t)187935104;          // temporal a_b

    // ---- spatial stage ----
    k_gemm<64, 96, 1, false, false><<<dim3(200, 1, NB), 192, 0, stream>>>(
        x, w_in_s, b_in_s, nullptr, nullptr, qkb);
    k_attn_s<<<dim3(NB * 3), 256, 0, stream>>>(qkb, alphas, att0s, attn);
    k_apply_s<<<dim3(96, NB), 256, 0, stream>>>(x, attn, big);
    k_gemm<192, 64, 1, true, true><<<dim3(200, 1, NB), 128, 0, stream>>>(
        big, w_out_s, b_out_s, bn_out_s, x, out);                    // y1 -> d_out
    k_gemm<64, 64, 1, true, true><<<dim3(200, 1, NB), 128, 0, stream>>>(
        out, w_ff_s, b_ff_s, bn_ff_s, x, yf);                        // y -> yf

    // ---- temporal stage ----
    k_gemm<64, 128, 1, false, false><<<dim3(200, 1, NB), 256, 0, stream>>>(
        yf, w_in_t, b_in_t, nullptr, nullptr, qkb);
    k_attn_t<<<dim3(8, 8, NB * 2), 32, 0, stream>>>(qkb, alphat_f, alphat_b, af, ab);
    k_apply_t<<<dim3(100, 8, NB * 2), 32, 0, stream>>>(yf, af, ab, big); // z(256ch)
    k_gemm<256, 64, 1, true, true><<<dim3(200, 1, NB), 128, 0, stream>>>(
        big, w_out_t, b_out_t, bn_out_t, yf, out);                   // z1 -> d_out
    k_gemm<64, 64, 1, true, true><<<dim3(200, 1, NB), 128, 0, stream>>>(
        out, w_ff_t, b_ff_t, bn_ff_t, yf, qkb);                      // z -> qkb

    // ---- TCN (7-tap, K = 64*7 = 448) fused bn + residual + lrelu ----
    k_gemm<64, 64, 7, true, true><<<dim3(200, 1, NB), 128, 0, stream>>>(
        qkb, w_tcn, b_tcn, bn_tcn, qkb, out);
}
